// BiLstmCellEncoder_34282428956782
// MI455X (gfx1250) — matrix operand order repaired and checked by reading, not compiled, not run on hardware
//
#include <hip/hip_runtime.h>
#include <hip/hip_bf16.h>

// ---------------------------------------------------------------------------
// BiLSTM encoder, B=128 S=200 D=H=512 L=2, bidirectional.
// Persistent per-layer kernel: 32 WGs per direction (16 h-columns each),
// grid.y = direction. Per step: gates = [h, x_t] @ [Whh, Wih]^T + b computed
// with v_wmma_f32_16x16x32_bf16 (weights LDS-resident, 128KB/WGP), cell state
// kept in the 16x16 f32 accumulator registers, grid barrier between steps.
// ---------------------------------------------------------------------------

typedef __attribute__((ext_vector_type(16))) __bf16 v16bf;
typedef __attribute__((ext_vector_type(8)))  float  v8f;

#define BB  128
#define SS  200
#define DD  512
#define HH  512
#define G4  (4*HH)      // 2048 gate rows
#define NWG 32          // workgroups per direction (32*16 = 512 = H columns)

union FragU { v16bf v; uint4 q[2]; };

__device__ __forceinline__ unsigned short f2bf(float f) {
    unsigned u = __float_as_uint(f);
    unsigned r = (u + 0x7FFFu + ((u >> 16) & 1u)) >> 16;   // round-to-nearest-even
    return (unsigned short)r;
}
__device__ __forceinline__ float sigm(float x)  { return 1.0f / (1.0f + __expf(-x)); }
__device__ __forceinline__ float tanh1(float x) { float e = __expf(2.0f * x); return 1.0f - 2.0f / (e + 1.0f); }

__device__ __forceinline__ void gridbar(unsigned* c, unsigned target) {
    __syncthreads();
    if (threadIdx.x == 0) {
        __threadfence();
        __hip_atomic_fetch_add(c, 1u, __ATOMIC_RELEASE, __HIP_MEMORY_SCOPE_AGENT);
        while (__hip_atomic_load(c, __ATOMIC_ACQUIRE, __HIP_MEMORY_SCOPE_AGENT) < target)
            __builtin_amdgcn_s_sleep(2);
        __threadfence();
    }
    __syncthreads();
}

// float32 -> bf16(ushort), 8 elements per thread
__global__ __launch_bounds__(256) void cvt_bf16x8(const float4* __restrict__ in,
                                                  uint4* __restrict__ out, int n8) {
    int i = blockIdx.x * 256 + threadIdx.x;
    if (i >= n8) return;
    float4 a = in[2 * i], b = in[2 * i + 1];
    union { unsigned short h[8]; uint4 q; } u;
    u.h[0] = f2bf(a.x); u.h[1] = f2bf(a.y); u.h[2] = f2bf(a.z); u.h[3] = f2bf(a.w);
    u.h[4] = f2bf(b.x); u.h[5] = f2bf(b.y); u.h[6] = f2bf(b.z); u.h[7] = f2bf(b.w);
    out[i] = u.q;
}

__global__ __launch_bounds__(256) void lstm_layer_kernel(
    const unsigned short* __restrict__ xinF,  // [BB,SS,DD] bf16
    const unsigned short* __restrict__ xinB,
    const unsigned short* __restrict__ WihF,  // [G4,DD] bf16
    const unsigned short* __restrict__ WihB,
    const unsigned short* __restrict__ WhhF,  // [G4,HH] bf16
    const unsigned short* __restrict__ WhhB,
    const float* __restrict__ biF, const float* __restrict__ bhF,
    const float* __restrict__ biB, const float* __restrict__ bhB,
    const long long* __restrict__ lens,
    unsigned short* __restrict__ hbuf,        // [2 dir][2 parity][BB][HH] bf16
    unsigned short* __restrict__ outBf,       // [2 dir][BB,SS,HH] bf16 masked (or null)
    float* __restrict__ outF,                 // [BB,SS,2*HH] f32 masked (or null)
    unsigned* __restrict__ ctr)               // [2] per-direction barrier counters
{
    const int dir  = blockIdx.y;
    const int jb   = blockIdx.x * 16;       // this WG's h-column base
    const int tid  = threadIdx.x;
    const int w    = tid >> 5;              // wave id: M-tile (rows 16w..16w+15)
    const int lane = tid & 31;
    const int nl   = lane & 15;
    const int half = lane >> 4;

    extern __shared__ char smem[];
    unsigned short* sWhh = (unsigned short*)smem;                   // [4][16][512] 64KB
    unsigned short* sWih = (unsigned short*)(smem + 64 * 1024);     // [4][16][512] 64KB
    float*          sBias = (float*)(smem + 128 * 1024);            // [64]
    int*            sLen  = (int*)(smem + 128 * 1024 + 256);        // [128]

    const unsigned short* xin = dir ? xinB : xinF;
    const unsigned short* Wih = dir ? WihB : WihF;
    const unsigned short* Whh = dir ? WhhB : WhhF;
    const float* bi = dir ? biB : biF;
    const float* bh = dir ? bhB : bhF;
    unsigned* bar = ctr + dir;
    unsigned short* hbase = hbuf + (size_t)dir * 2 * BB * HH;

    // --- preload weight slices into LDS: LDS row g*16+n <- global row g*HH+jb+n
    for (int idx = tid; idx < 64 * 64; idx += 256) {
        int row = idx >> 6, ch = idx & 63;               // 64 chunks * 8 halfs = 512
        int g = row >> 4, n = row & 15;
        size_t grow = (size_t)(g * HH + jb + n);
        *(uint4*)(sWhh + row * HH + ch * 8) = *(const uint4*)(Whh + grow * HH + ch * 8);
        *(uint4*)(sWih + row * DD + ch * 8) = *(const uint4*)(Wih + grow * DD + ch * 8);
    }
    if (tid < 64)  sBias[tid] = bi[(tid >> 4) * HH + jb + (tid & 15)]
                              + bh[(tid >> 4) * HH + jb + (tid & 15)];
    if (tid < BB)  sLen[tid] = (int)lens[tid];

    // --- zero h (parity 0) for our column slice, all batch rows
    for (int i = tid; i < BB * 16; i += 256)
        hbase[(i >> 4) * HH + jb + (i & 15)] = 0;
    __syncthreads();
    unsigned phase = 1;
    gridbar(bar, NWG * phase); phase++;

    // cell state tile lives in registers: c[r] for row m=r+8*half, col n=nl
    v8f cst = {0.f, 0.f, 0.f, 0.f, 0.f, 0.f, 0.f, 0.f};

    const int brow = w * 16 + nl;                        // A-fragment row (batch index)

    for (int t = 0; t < SS; ++t) {
        const int tt = dir ? (SS - 1 - t) : t;
        const unsigned short* hsrc = hbase + (t & 1) * BB * HH;
        unsigned short*       hdst = hbase + ((t + 1) & 1) * BB * HH;

        v8f a0 = {0.f,0.f,0.f,0.f,0.f,0.f,0.f,0.f};
        v8f a1 = a0, a2 = a0, a3 = a0;

        const unsigned short* hrow = hsrc + (size_t)brow * HH;
        const unsigned short* xrow = xin + ((size_t)brow * SS + tt) * DD;
        if (t + 1 < SS) {
            const int nt = dir ? (SS - 2 - t) : (t + 1);
            __builtin_prefetch(xin + ((size_t)brow * SS + nt) * DD, 0, 1);
        }

        // ---- gates += h @ Whh^T  (K = 512, 16 chunks of 32)
        #pragma unroll 4
        for (int kc = 0; kc < 16; ++kc) {
            FragU a;   // A 16x32 bf16, this code's k order: 16 contiguous K at K=16*half
            a.q[0] = *(const uint4*)(hrow + kc * 32 + half * 16);
            a.q[1] = *(const uint4*)(hrow + kc * 32 + half * 16 + 8);
            const unsigned short* p = sWhh + nl * HH + kc * 32 + half * 16;
            FragU b0, b1, b2, b3;  // B 32x16: 16 contiguous K from weight row n
            b0.q[0] = *(const uint4*)(p);               b0.q[1] = *(const uint4*)(p + 8);
            b1.q[0] = *(const uint4*)(p + 16 * HH);     b1.q[1] = *(const uint4*)(p + 16 * HH + 8);
            b2.q[0] = *(const uint4*)(p + 32 * HH);     b2.q[1] = *(const uint4*)(p + 32 * HH + 8);
            b3.q[0] = *(const uint4*)(p + 48 * HH);     b3.q[1] = *(const uint4*)(p + 48 * HH + 8);
            a0 = __builtin_amdgcn_wmma_f32_16x16x32_bf16(false, a.v, false, b0.v, (short)0, a0, false, false);
            a1 = __builtin_amdgcn_wmma_f32_16x16x32_bf16(false, a.v, false, b1.v, (short)0, a1, false, false);
            a2 = __builtin_amdgcn_wmma_f32_16x16x32_bf16(false, a.v, false, b2.v, (short)0, a2, false, false);
            a3 = __builtin_amdgcn_wmma_f32_16x16x32_bf16(false, a.v, false, b3.v, (short)0, a3, false, false);
        }
        // ---- gates += x_t @ Wih^T
        #pragma unroll 4
        for (int kc = 0; kc < 16; ++kc) {
            FragU a;
            a.q[0] = *(const uint4*)(xrow + kc * 32 + half * 16);
            a.q[1] = *(const uint4*)(xrow + kc * 32 + half * 16 + 8);
            const unsigned short* p = sWih + nl * DD + kc * 32 + half * 16;
            FragU b0, b1, b2, b3;
            b0.q[0] = *(const uint4*)(p);               b0.q[1] = *(const uint4*)(p + 8);
            b1.q[0] = *(const uint4*)(p + 16 * DD);     b1.q[1] = *(const uint4*)(p + 16 * DD + 8);
            b2.q[0] = *(const uint4*)(p + 32 * DD);     b2.q[1] = *(const uint4*)(p + 32 * DD + 8);
            b3.q[0] = *(const uint4*)(p + 48 * DD);     b3.q[1] = *(const uint4*)(p + 48 * DD + 8);
            a0 = __builtin_amdgcn_wmma_f32_16x16x32_bf16(false, a.v, false, b0.v, (short)0, a0, false, false);
            a1 = __builtin_amdgcn_wmma_f32_16x16x32_bf16(false, a.v, false, b1.v, (short)0, a1, false, false);
            a2 = __builtin_amdgcn_wmma_f32_16x16x32_bf16(false, a.v, false, b2.v, (short)0, a2, false, false);
            a3 = __builtin_amdgcn_wmma_f32_16x16x32_bf16(false, a.v, false, b3.v, (short)0, a3, false, false);
        }

        // ---- elementwise LSTM cell; C/D layout: row = r + 8*half, col = nl
        #pragma unroll
        for (int r = 0; r < 8; ++r) {
            const int b = w * 16 + half * 8 + r;
            float iv = sigm (a0[r] + sBias[     nl]);
            float fv = sigm (a1[r] + sBias[16 + nl]);
            float gv = tanh1(a2[r] + sBias[32 + nl]);
            float ov = sigm (a3[r] + sBias[48 + nl]);
            float cv = fv * cst[r] + iv * gv;
            cst[r] = cv;
            float hv = ov * tanh1(cv);
            hdst[(size_t)b * HH + jb + nl] = f2bf(hv);            // recurrence (unmasked)
            float mv = (tt < sLen[b]) ? hv : 0.0f;                // layer output (masked)
            if (outBf) outBf[(size_t)dir * BB * SS * HH + ((size_t)b * SS + tt) * HH + jb + nl] = f2bf(mv);
            if (outF)  outF[((size_t)b * SS + tt) * (2 * HH) + dir * HH + jb + nl] = mv;
        }
        gridbar(bar, NWG * phase); phase++;
    }
}

extern "C" void kernel_launch(void* const* d_in, const int* in_sizes, int n_in,
                              void* d_out, int out_size, void* d_ws, size_t ws_size,
                              hipStream_t stream) {
    const float*     x    = (const float*)d_in[0];
    const long long* lens = (const long long*)d_in[1];
    const float* WihF = (const float*)d_in[2];
    const float* WhhF = (const float*)d_in[3];
    const float* biF  = (const float*)d_in[4];
    const float* bhF  = (const float*)d_in[5];
    const float* WihB = (const float*)d_in[6];
    const float* WhhB = (const float*)d_in[7];
    const float* biB  = (const float*)d_in[8];
    const float* bhB  = (const float*)d_in[9];
    float* out = (float*)d_out;

    // ---- workspace layout (bf16 stored as ushort) ----
    char* ws = (char*)d_ws;
    size_t off = 0;
    unsigned* ctr = (unsigned*)(ws + off); off += 256;                    // 4 counters
    unsigned short* xbf  = (unsigned short*)(ws + off); off += (size_t)BB * SS * DD * 2;
    unsigned short* wihF = (unsigned short*)(ws + off); off += (size_t)2 * G4 * DD * 2;
    unsigned short* wihB = (unsigned short*)(ws + off); off += (size_t)2 * G4 * DD * 2;
    unsigned short* whhF = (unsigned short*)(ws + off); off += (size_t)2 * G4 * HH * 2;
    unsigned short* whhB = (unsigned short*)(ws + off); off += (size_t)2 * G4 * HH * 2;
    unsigned short* hbuf = (unsigned short*)(ws + off); off += (size_t)2 * 2 * BB * HH * 2;
    unsigned short* l0o  = (unsigned short*)(ws + off); off += (size_t)2 * BB * SS * HH * 2;

    hipMemsetAsync(ctr, 0, 256, stream);

    // ---- fp32 -> bf16 conversions ----
    int n8;
    n8 = BB * SS * DD / 8;
    cvt_bf16x8<<<(n8 + 255) / 256, 256, 0, stream>>>((const float4*)x, (uint4*)xbf, n8);
    n8 = 2 * G4 * DD / 8;
    cvt_bf16x8<<<(n8 + 255) / 256, 256, 0, stream>>>((const float4*)WihF, (uint4*)wihF, n8);
    cvt_bf16x8<<<(n8 + 255) / 256, 256, 0, stream>>>((const float4*)WihB, (uint4*)wihB, n8);
    n8 = 2 * G4 * HH / 8;
    cvt_bf16x8<<<(n8 + 255) / 256, 256, 0, stream>>>((const float4*)WhhF, (uint4*)whhF, n8);
    cvt_bf16x8<<<(n8 + 255) / 256, 256, 0, stream>>>((const float4*)WhhB, (uint4*)whhB, n8);

    const size_t smem = 128 * 1024 + 256 + 512;   // Whh + Wih slices, bias, lens
    dim3 grid(NWG, 2);

    // ---- layer 0: input x (both dirs), outputs masked bf16 for layer 1 ----
    lstm_layer_kernel<<<grid, 256, smem, stream>>>(
        xbf, xbf,
        wihF, wihB, whhF, whhB,
        biF, bhF, biB, bhB, lens,
        hbuf, l0o, nullptr, ctr);

    // ---- layer 1: inputs are layer-0 masked outputs, writes d_out ----
    lstm_layer_kernel<<<grid, 256, smem, stream>>>(
        l0o, l0o + (size_t)BB * SS * HH,
        wihF + (size_t)G4 * DD, wihB + (size_t)G4 * DD,
        whhF + (size_t)G4 * HH, whhB + (size_t)G4 * HH,
        biF + G4, bhF + G4, biB + G4, bhB + G4, lens,
        hbuf, nullptr, out, ctr + 2);

    (void)in_sizes; (void)n_in; (void)out_size; (void)ws_size;
}
